// AttentionDecoder_429496729625
// MI455X (gfx1250) — matrix operand order repaired and checked by reading, not compiled, not run on hardware
//
#include <hip/hip_runtime.h>
#include <hip/hip_bf16.h>

// ---------------------------------------------------------------------------
// AttentionDecoder forward for MI455X (gfx1250, wave32, WMMA).
// - encoder @ Wa1[:256] is step-invariant -> hoisted out of the 16-step loop,
//   stored bf16 (48MB) so per-step attention working set is L2-resident.
// - All GEMMs: v_wmma_f32_16x16x32_bf16, f32->bf16 converted on the fly.
// - GEMM templated on <FLAGS, NT>: branch-free K-loop; NT adjacent 16-wide
//   N-tiles per wave so the A operand is loaded once and fed to NT
//   back-to-back WMMAs (A-traffic /NT; critical for the M=186880 GEMM).
// ---------------------------------------------------------------------------

typedef __attribute__((ext_vector_type(16))) __bf16 v16bf;
typedef __attribute__((ext_vector_type(8)))  float  v8f;

#define BATCH  512
#define TSTEPS 365
#define DDIM   256
#define NREC   256
#define NF     16

#define FLAG_RELU    1
#define FLAG_TRANSB  2
#define FLAG_ACC     4
#define FLAG_BF16OUT 8
#define FLAG_BIAS    16

// Generic GEMM: C[M,N] = A[M,K] @ B[K,N] (+bias)(+C)(relu)(bf16 store).
// Wave w = blockIdx.x*8+wave owns M-tile tm; covers NT N-tiles starting at
// blockIdx.y*NT*16. grid = (ceil(M/16/8), N/(NT*16)).
// K multiple of 32; M,N multiples of 16; N multiple of NT*16.
template <int FLAGS, int NT>
__global__ void wmma_gemm_bf16(const float* __restrict__ A,
                               const float* __restrict__ Bm,
                               const float* __restrict__ bias,
                               void* __restrict__ Cv,
                               int M, int K, int lda, int ldb, int ldc)
{
    const int lane = threadIdx.x & 31;
    const int wave = threadIdx.x >> 5;
    const long tm  = (long)(blockIdx.x * 8 + wave) << 4;
    const int  tn0 = blockIdx.y * (NT << 4);
    if (tm >= M) return;                 // wave-uniform: EXEC stays all-ones
    const int h   = lane >> 4;           // half-wave selector (K-range)
    const int l15 = lane & 15;

    // A: contiguous k = 16h + s order (B matches). Fold h*16 into the row pointer; per-VGPR pair
    // offset is then k + 2*v, always 8B-aligned -> float2.
    const float* arow = A + (tm + l15) * (long)lda + (h << 4);
    v8f acc[NT] = {};

    if constexpr ((FLAGS & FLAG_TRANSB) != 0) {
        // B[k][n] = Bm[n*ldb + k]; fold h*16 in -> contiguous float2 loads.
        const float* brow = Bm + (long)(tn0 + l15) * ldb + (h << 4);
        for (int k = 0; k < K; k += 32) {
            __builtin_prefetch(arow + k + 64, 0, 1);
            v16bf a;
#pragma unroll
            for (int v = 0; v < 8; ++v) {
                float2 av = *(const float2*)(arow + k + (v << 1));
                a[2 * v]     = (__bf16)av.x;
                a[2 * v + 1] = (__bf16)av.y;
            }
#pragma unroll
            for (int t = 0; t < NT; ++t) {
                const float* bt = brow + (long)(t << 4) * ldb + k;
                v16bf b;
#pragma unroll
                for (int v = 0; v < 8; ++v) {
                    float2 bv = *(const float2*)(bt + (v << 1));
                    b[2 * v]     = (__bf16)bv.x;
                    b[2 * v + 1] = (__bf16)bv.y;
                }
                acc[t] = __builtin_amdgcn_wmma_f32_16x16x32_bf16(
                    false, a, false, b, (short)0, acc[t], false, false);
            }
        }
    } else {
        // B row-major: column tn0+t*16+l15, rows k+16h+2v (+1): b32 loads.
        const float* bcol = Bm + tn0 + l15 + (long)(h << 4) * ldb;
        for (int k = 0; k < K; k += 32) {
            __builtin_prefetch(arow + k + 64, 0, 1);
            v16bf a;
#pragma unroll
            for (int v = 0; v < 8; ++v) {
                float2 av = *(const float2*)(arow + k + (v << 1));
                a[2 * v]     = (__bf16)av.x;
                a[2 * v + 1] = (__bf16)av.y;
            }
            const float* bk = bcol + (long)k * ldb;
#pragma unroll
            for (int t = 0; t < NT; ++t) {
                v16bf b;
#pragma unroll
                for (int v = 0; v < 8; ++v) {
                    b[2 * v]     = (__bf16)bk[(long)(v << 1) * ldb + (t << 4)];
                    b[2 * v + 1] = (__bf16)bk[(long)((v << 1) + 1) * ldb + (t << 4)];
                }
                acc[t] = __builtin_amdgcn_wmma_f32_16x16x32_bf16(
                    false, a, false, b, (short)0, acc[t], false, false);
            }
        }
    }

    // Epilogue. C/D layout: VGPR r, half h -> M = r + 8h, N = l15.
#pragma unroll
    for (int t = 0; t < NT; ++t) {
        const int col = tn0 + (t << 4) + l15;
        float bv = 0.f;
        if constexpr ((FLAGS & FLAG_BIAS) != 0) bv = bias[col];
        const long base = (tm + (h << 3)) * (long)ldc + col;
#pragma unroll
        for (int r = 0; r < 8; ++r) {
            float v = acc[t][r] + bv;
            if constexpr ((FLAGS & FLAG_ACC) != 0)
                v += ((const float*)Cv)[base + (long)r * ldc];
            if constexpr ((FLAGS & FLAG_RELU) != 0)
                v = fmaxf(v, 0.f);
            if constexpr ((FLAGS & FLAG_BF16OUT) != 0)
                ((__hip_bfloat16*)Cv)[base + (long)r * ldc] = __float2bfloat16(v);
            else
                ((float*)Cv)[base + (long)r * ldc] = v;
        }
    }
}

template <int FLAGS, int NT>
static void gemm_launch(hipStream_t stream,
                        const float* A, const float* Bm, const float* bias,
                        void* C, int M, int N, int K, int lda, int ldb, int ldc)
{
    dim3 grid(((M >> 4) + 7) >> 3, N / (NT << 4));
    wmma_gemm_bf16<FLAGS, NT><<<grid, 256, 0, stream>>>(A, Bm, bias, C,
                                                        M, K, lda, ldb, ldc);
}

// Build cat = [emb_store, emb_item, emb_family, state_h, state_c]  (512 x 608)
__global__ void gather_cat_kernel(const int* __restrict__ xcat,
                                  const float* __restrict__ emb_store,
                                  const float* __restrict__ emb_item,
                                  const float* __restrict__ emb_family,
                                  const float* __restrict__ state_h,
                                  const float* __restrict__ state_c,
                                  float* __restrict__ cat)
{
    int idx = blockIdx.x * blockDim.x + threadIdx.x;
    if (idx >= BATCH * 608) return;
    int b = idx / 608, col = idx - b * 608;
    float v;
    if (col < 16)        v = emb_store [xcat[b * 3 + 0] * 16 + col];
    else if (col < 80)   v = emb_item  [xcat[b * 3 + 1] * 64 + (col - 16)];
    else if (col < 96)   v = emb_family[xcat[b * 3 + 2] * 16 + (col - 80)];
    else if (col < 352)  v = state_h[b * 256 + (col - 96)];
    else                 v = state_c[b * 256 + (col - 352)];
    cat[idx] = v;
}

// scores[b*T+t] = relu(enc_part[t*B+b,:] + st_part[b,:]) . Wa2 + ba2
// One wave per (t,b); lane-strided dot + shuffle reduction.
__global__ void scores_kernel(const __hip_bfloat16* __restrict__ encpart,
                              const float* __restrict__ stpart,
                              const float* __restrict__ Wa2,
                              const float* __restrict__ ba2,
                              float* __restrict__ scores)
{
    int wid  = (blockIdx.x * blockDim.x + threadIdx.x) >> 5;  // wave-uniform
    int lane = threadIdx.x & 31;
    if (wid >= TSTEPS * BATCH) return;
    int t = wid >> 9, b = wid & (BATCH - 1);                  // BATCH = 512
    const __hip_bfloat16* ep = encpart + (long)wid * 128;
    const float* sp = stpart + b * 128;
    float acc = 0.f;
#pragma unroll
    for (int u = 0; u < 4; ++u) {
        int j = lane + (u << 5);
        float v = __bfloat162float(ep[j]) + sp[j];
        acc += fmaxf(v, 0.f) * Wa2[j];
    }
    for (int off = 16; off; off >>= 1) acc += __shfl_xor(acc, off, 32);
    if (lane == 0) scores[(long)b * TSTEPS + t] = acc + ba2[0];
}

// Per-b softmax over T and weighted sum of encoder rows -> inp[b, 0:256]
__global__ void softmax_weighted_kernel(const float* __restrict__ scores,
                                        const float* __restrict__ encoder,
                                        float* __restrict__ inp)
{
    __shared__ float sw[TSTEPS];
    __shared__ float red[256];
    int b = blockIdx.x, tid = threadIdx.x;
    const float* sc = scores + (long)b * TSTEPS;

    float lm = -3.4e38f;
    for (int t = tid; t < TSTEPS; t += 256) lm = fmaxf(lm, sc[t]);
    red[tid] = lm; __syncthreads();
    for (int s = 128; s; s >>= 1) {
        if (tid < s) red[tid] = fmaxf(red[tid], red[tid + s]);
        __syncthreads();
    }
    float mx = red[0]; __syncthreads();

    float ls = 0.f;
    for (int t = tid; t < TSTEPS; t += 256) {
        float e = __expf(sc[t] - mx); sw[t] = e; ls += e;
    }
    red[tid] = ls; __syncthreads();
    for (int s = 128; s; s >>= 1) {
        if (tid < s) red[tid] += red[tid + s];
        __syncthreads();
    }
    float inv = 1.f / red[0]; __syncthreads();
    for (int t = tid; t < TSTEPS; t += 256) sw[t] *= inv;
    __syncthreads();

    int j = tid;  // 0..255 = feature
    float acc = 0.f;
    for (int t = 0; t < TSTEPS; ++t)
        acc += sw[t] * encoder[((long)t * BATCH + b) * DDIM + j];
    inp[b * DDIM + j] = acc;
}

// LSTM cell: gates (B x 1024) + c -> h_new, c_new (hc layout: [h | c] per row)
__global__ void lstm_kernel(const float* __restrict__ gates,
                            float* __restrict__ hc,
                            float* __restrict__ outs_step)
{
    int idx = blockIdx.x * blockDim.x + threadIdx.x;   // B*256
    int b = idx >> 8, j = idx & 255;
    float gi = gates[b * 1024 + j];
    float gf = gates[b * 1024 + 256 + j];
    float gg = gates[b * 1024 + 512 + j];
    float go = gates[b * 1024 + 768 + j];
    float c  = hc[b * 512 + 256 + j];
    float si = 1.f / (1.f + __expf(-gi));
    float sf = 1.f / (1.f + __expf(-gf));
    float so = 1.f / (1.f + __expf(-go));
    float cn = sf * c + si * tanhf(gg);
    float hn = so * tanhf(cn);
    hc[b * 512 + j]       = hn;
    hc[b * 512 + 256 + j] = cn;
    outs_step[b * 256 + j] = hn;
}

// Time-distributed head: out[s*B+b] = relu(outs[s,b,:] @ Wt1 + bt1) @ Wt2 + bt2
__global__ void td_kernel(const float* __restrict__ outs,
                          const float* __restrict__ Wt1,
                          const float* __restrict__ bt1,
                          const float* __restrict__ Wt2,
                          const float* __restrict__ bt2,
                          float* __restrict__ out)
{
    int wid  = (blockIdx.x * blockDim.x + threadIdx.x) >> 5;
    int lane = threadIdx.x & 31;
    if (wid >= NF * BATCH) return;
    const float* x = outs + (long)wid * NREC;
    float acc = 0.f;
    for (int hu = lane; hu < 128; hu += 32) {
        float d = bt1[hu];
        for (int k = 0; k < NREC; ++k) d += x[k] * Wt1[k * 128 + hu];
        acc += fmaxf(d, 0.f) * Wt2[hu];
    }
    for (int off = 16; off; off >>= 1) acc += __shfl_xor(acc, off, 32);
    if (lane == 0) out[wid] = acc + bt2[0];
}

extern "C" void kernel_launch(void* const* d_in, const int* in_sizes, int n_in,
                              void* d_out, int out_size, void* d_ws, size_t ws_size,
                              hipStream_t stream)
{
    const int*   xcat      = (const int*)  d_in[0];
    const float* state_h   = (const float*)d_in[1];
    const float* state_c   = (const float*)d_in[2];
    const float* encoder   = (const float*)d_in[3];
    const float* emb_store = (const float*)d_in[4];
    const float* emb_item  = (const float*)d_in[5];
    const float* emb_family= (const float*)d_in[6];
    const float* Wc1 = (const float*)d_in[7];  const float* bc1 = (const float*)d_in[8];
    const float* Wc2 = (const float*)d_in[9];  const float* bc2 = (const float*)d_in[10];
    const float* Wc3 = (const float*)d_in[11]; const float* bc3 = (const float*)d_in[12];
    const float* Wa1 = (const float*)d_in[13]; const float* ba1 = (const float*)d_in[14];
    const float* Wa2 = (const float*)d_in[15]; const float* ba2 = (const float*)d_in[16];
    const float* Wt1 = (const float*)d_in[17]; const float* bt1 = (const float*)d_in[18];
    const float* Wt2 = (const float*)d_in[19]; const float* bt2 = (const float*)d_in[20];
    const float* Wih = (const float*)d_in[21];
    const float* Whh = (const float*)d_in[22];
    const float* bih = (const float*)d_in[23];
    const float* bhh = (const float*)d_in[24];
    float* out = (float*)d_out;

    // ---- workspace bump allocator (~64 MB total) ----
    char* wsb = (char*)d_ws;
    size_t off = 0;
    auto walloc = [&](size_t bytes) -> void* {
        void* p = wsb + off;
        off = (off + bytes + 255) & ~(size_t)255;
        return p;
    };
    const long MENC = (long)TSTEPS * BATCH;                     // 186880
    float*          cat     = (float*)walloc((size_t)BATCH * 608 * 4);
    float*          ct1     = (float*)walloc((size_t)BATCH * 512 * 4);
    float*          ct2     = (float*)walloc((size_t)BATCH * 384 * 4);
    float*          hc      = (float*)walloc((size_t)BATCH * 512 * 4);  // [h|c]
    __hip_bfloat16* encpart = (__hip_bfloat16*)walloc((size_t)MENC * 128 * 2);
    float*          stpart  = (float*)walloc((size_t)BATCH * 128 * 4);
    float*          scores  = (float*)walloc((size_t)BATCH * TSTEPS * 4);
    float*          inp     = (float*)walloc((size_t)BATCH * DDIM * 4);
    float*          gates   = (float*)walloc((size_t)BATCH * 1024 * 4);
    float*          outs    = (float*)walloc((size_t)NF * BATCH * NREC * 4);
    (void)ws_size; (void)n_in; (void)in_sizes; (void)out_size;

    // ---- context MLP: cat -> relu(Wc1) -> relu(Wc2) -> Wc3 = [h0|c0] ----
    gather_cat_kernel<<<(BATCH * 608 + 255) / 256, 256, 0, stream>>>(
        xcat, emb_store, emb_item, emb_family, state_h, state_c, cat);
    gemm_launch<FLAG_RELU | FLAG_BIAS, 1>(stream, cat, Wc1, bc1, ct1,
                                          BATCH, 512, 608, 608, 512, 512);
    gemm_launch<FLAG_RELU | FLAG_BIAS, 1>(stream, ct1, Wc2, bc2, ct2,
                                          BATCH, 384, 512, 512, 384, 384);
    gemm_launch<FLAG_BIAS, 1>(stream, ct2, Wc3, bc3, hc,
                              BATCH, 512, 384, 384, 512, 512);

    // ---- hoisted step-invariant projection: encoder @ Wa1[:256] (bf16 out) ----
    // NT=8: one wave covers the whole N=128 strip -> A read exactly once.
    gemm_launch<FLAG_BF16OUT, 8>(stream, encoder, Wa1, nullptr, encpart,
                                 (int)MENC, 128, 256, 256, 128, 128);

    // ---- 16 decode steps ----
    for (int step = 0; step < NF; ++step) {
        // st_part = [h|c] @ Wa1[256:] + ba1
        gemm_launch<FLAG_BIAS, 1>(stream, hc, Wa1 + 256 * 128, ba1, stpart,
                                  BATCH, 128, 512, 512, 128, 128);
        scores_kernel<<<(TSTEPS * BATCH) / 8, 256, 0, stream>>>(
            encpart, stpart, Wa2, ba2, scores);
        softmax_weighted_kernel<<<BATCH, 256, 0, stream>>>(scores, encoder, inp);
        // gates = inp @ Wih^T + bih  +  h @ Whh^T + bhh
        gemm_launch<FLAG_TRANSB | FLAG_BIAS, 4>(stream, inp, Wih, bih, gates,
                                                BATCH, 1024, 256, 256, 256, 1024);
        gemm_launch<FLAG_TRANSB | FLAG_ACC | FLAG_BIAS, 4>(stream, hc, Whh, bhh, gates,
                                                           BATCH, 1024, 256, 512, 256, 1024);
        lstm_kernel<<<BATCH, 256, 0, stream>>>(gates, hc,
                                               outs + (size_t)step * BATCH * NREC);
    }

    // ---- time-distributed head ----
    td_kernel<<<(NF * BATCH) / 8, 256, 0, stream>>>(outs, Wt1, bt1, Wt2, bt2, out);
}
